// FourierCrossAttentionW_30039001268331
// MI455X (gfx1250) — compile-verified, matrix operand order checked by reading, not run on hardware
//
#include <hip/hip_runtime.h>

// ---------------------------------------------------------------------------
// FourierCrossAttentionW pipeline for MI455X (gfx1250, wave32, WMMA bf16)
//
// All matmuls are NT GEMMs (C[m,n] = sum_k A[m,k]*B[n,k]) run through
// v_wmma_f32_16x16x32_bf16. rfft/irfft are folded into DFT matmuls.
// Accumulators are named v8f registers (no runtime-indexed arrays -> no
// scratch spill); LDS tiles are double-buffered (one barrier per k-step).
// ---------------------------------------------------------------------------

typedef __bf16 bf16;
typedef __attribute__((ext_vector_type(8)))  __bf16 bf16x8;
typedef __attribute__((ext_vector_type(16))) __bf16 v16bf;
typedef __attribute__((ext_vector_type(8)))  float  v8f;
typedef __attribute__((ext_vector_type(4)))  float  f32x4;

#define BM 64
#define BN 64
#define BK 32
#define LDS_K 40   // padded row stride (bf16 elems); 40*2=80B keeps 16B alignment

// ---------------------------------------------------------------------------
// Generic batched NT GEMM:
//   C[z,m,n] = a1 * sum_k A[z,m,k]B[z,n,k] + a2 * sum_k A2[z,m,k]B2[z,n,k]
//              + bias[n] + addin[z,m,n]
// A/B element type selected by AF32/BF32 (converted to bf16 in LDS).
// Output f32 or bf16; optional transposed store C[z,n,m].
// M,N,K(,K2) must be multiples of 64/64/32 (true for every launch below).
// ---------------------------------------------------------------------------
template<bool AF32, bool BF32, bool OUTBF16, bool OUTTRANS>
__global__ __launch_bounds__(256)
void gemm_nt(const void* __restrict__ Av, const void* __restrict__ Bv,
             const void* __restrict__ A2v, const void* __restrict__ B2v,
             const float* __restrict__ bias, const float* __restrict__ addin,
             void* __restrict__ Cv,
             int K, int K2,
             int lda, int ldb, int ldc,
             long sA, long sB, long sC, long sAdd,
             float a1, float a2)
{
    __shared__ __align__(16) bf16 lA[2][BM * LDS_K];
    __shared__ __align__(16) bf16 lB[2][BN * LDS_K];

    const int tid  = threadIdx.x;
    const int lane = tid & 31;
    const int wave = tid >> 5;
    const int z    = blockIdx.z;
    const int m0   = blockIdx.y * BM;
    const int n0   = blockIdx.x * BN;

    // 8 waves -> 4 (M) x 2 (N) arrangement; each wave does two 16-wide N tiles
    const int wm     = (wave >> 1) * 16;
    const int wnBase = (wave & 1) * 32;

    // tile loader: thread t loads 8 contiguous K elems of row (t>>2)
    const int lr = tid >> 2;
    const int lc = (tid & 3) * 8;

    // fragment decode (16-bit A/B layout, ISA 7.12.2)
    const int fr  = lane & 15;                // M (or N) index within tile
    const int kg  = (lane >> 4) ? 8 : 0;      // K sub-group offset
    const int rhi = (lane >> 4) << 3;         // C/D row offset for hi lanes

    const v8f vzero = {0.f,0.f,0.f,0.f,0.f,0.f,0.f,0.f};
    v8f acc10 = vzero, acc11 = vzero;         // pair 1, two N sub-tiles
    v8f acc20 = vzero, acc21 = vzero;         // pair 2

    // ---- tile staging (global -> LDS bf16) --------------------------------
    auto stageA = [&](const void* Ap, int kb, int buf) {
        if constexpr (AF32) {
            const float* Ar = (const float*)Ap + (long)z*sA + (long)(m0+lr)*lda + (kb+lc);
            f32x4 v0 = *(const f32x4*)Ar;
            f32x4 v1 = *(const f32x4*)(Ar + 4);
            __builtin_prefetch(Ar + BK, 0, 1);
            bf16x8 h;
            #pragma unroll
            for (int i = 0; i < 4; ++i) { h[i] = (bf16)v0[i]; h[i+4] = (bf16)v1[i]; }
            *(bf16x8*)&lA[buf][lr*LDS_K + lc] = h;
        } else {
            const bf16* Ar = (const bf16*)Ap + (long)z*sA + (long)(m0+lr)*lda + (kb+lc);
            __builtin_prefetch(Ar + BK, 0, 1);
            *(bf16x8*)&lA[buf][lr*LDS_K + lc] = *(const bf16x8*)Ar;
        }
    };
    auto stageB = [&](const void* Bp, int kb, int buf) {
        if constexpr (BF32) {
            const float* Br = (const float*)Bp + (long)z*sB + (long)(n0+lr)*ldb + (kb+lc);
            f32x4 v0 = *(const f32x4*)Br;
            f32x4 v1 = *(const f32x4*)(Br + 4);
            __builtin_prefetch(Br + BK, 0, 1);
            bf16x8 h;
            #pragma unroll
            for (int i = 0; i < 4; ++i) { h[i] = (bf16)v0[i]; h[i+4] = (bf16)v1[i]; }
            *(bf16x8*)&lB[buf][lr*LDS_K + lc] = h;
        } else {
            const bf16* Br = (const bf16*)Bp + (long)z*sB + (long)(n0+lr)*ldb + (kb+lc);
            __builtin_prefetch(Br + BK, 0, 1);
            *(bf16x8*)&lB[buf][lr*LDS_K + lc] = *(const bf16x8*)Br;
        }
    };

    // ---- double-buffered k-loop over one A/B product ----------------------
    auto kloop = [&](const void* Ap, const void* Bp, int Kp, v8f& aT0, v8f& aT1) {
        __syncthreads();                      // LDS reusable (prev compute done)
        stageA(Ap, 0, 0);
        stageB(Bp, 0, 0);
        int buf = 0;
        for (int kb = 0; kb < Kp; kb += BK, buf ^= 1) {
            __syncthreads();                  // tile[buf] visible; tile[buf^1] free
            if (kb + BK < Kp) {               // overlap next-tile loads with WMMA
                stageA(Ap, kb + BK, buf ^ 1);
                stageB(Bp, kb + BK, buf ^ 1);
            }
            // A fragment: row wm+fr, K pattern {kg..kg+7, 16+kg..16+kg+7}
            bf16x8 alo = *(const bf16x8*)&lA[buf][(wm+fr)*LDS_K + kg];
            bf16x8 ahi = *(const bf16x8*)&lA[buf][(wm+fr)*LDS_K + 16 + kg];
            v16bf af;
            #pragma unroll
            for (int i = 0; i < 8; ++i) { af[i] = alo[i]; af[i+8] = ahi[i]; }
            // B fragment / WMMA, N sub-tile 0
            {
                bf16x8 blo = *(const bf16x8*)&lB[buf][(wnBase+fr)*LDS_K + kg];
                bf16x8 bhi = *(const bf16x8*)&lB[buf][(wnBase+fr)*LDS_K + 16 + kg];
                v16bf bfr;
                #pragma unroll
                for (int i = 0; i < 8; ++i) { bfr[i] = blo[i]; bfr[i+8] = bhi[i]; }
                aT0 = __builtin_amdgcn_wmma_f32_16x16x32_bf16(
                    false, af, false, bfr, (short)0, aT0, false, false);
            }
            // B fragment / WMMA, N sub-tile 1
            {
                bf16x8 blo = *(const bf16x8*)&lB[buf][(wnBase+16+fr)*LDS_K + kg];
                bf16x8 bhi = *(const bf16x8*)&lB[buf][(wnBase+16+fr)*LDS_K + 16 + kg];
                v16bf bfr;
                #pragma unroll
                for (int i = 0; i < 8; ++i) { bfr[i] = blo[i]; bfr[i+8] = bhi[i]; }
                aT1 = __builtin_amdgcn_wmma_f32_16x16x32_bf16(
                    false, af, false, bfr, (short)0, aT1, false, false);
            }
        }
    };

    kloop(Av, Bv, K, acc10, acc11);
    if (A2v != nullptr) kloop(A2v, B2v, K2, acc20, acc21);

    // ---- epilogue: combine pairs, bias/add-in, store ----------------------
    auto store = [&](const v8f& p1, const v8f& p2, int nn) {
        const int col = n0 + nn + fr;
        #pragma unroll
        for (int v = 0; v < 8; ++v) {
            const int row = m0 + wm + v + rhi;
            float val = a1 * p1[v] + a2 * p2[v];
            if (bias)  val += bias[col];
            if (addin) val += addin[(long)z*sAdd + (long)row*ldc + col];
            const long cidx = OUTTRANS ? ((long)z*sC + (long)col*ldc + row)
                                       : ((long)z*sC + (long)row*ldc + col);
            if (OUTBF16) ((bf16*)Cv)[cidx] = (bf16)val;
            else         ((float*)Cv)[cidx] = val;
        }
    };
    store(acc10, acc20, wnBase);
    store(acc11, acc21, wnBase + 16);
}

// ---------------------------------------------------------------------------
// DFT / inverse-DFT tables (irfft weights folded in; Im(X0) dropped like
// numpy.fft.irfft; bin N/2 is zero since modes=64 < 65).
// ---------------------------------------------------------------------------
__global__ void init_tables(bf16* dftRe, bf16* dftIm, bf16* idftC, bf16* idftS)
{
    const int i = blockIdx.x * 256 + threadIdx.x;
    if (i >= 8192) return;
    const float TWO_PI = 6.28318530717958647692f;
    {   // forward: [m][n], m<64, n<128 ; X_m = sum_n x_n e^{-2pi i m n/128}
        const int m = i >> 7, n = i & 127;
        const float th = TWO_PI * (float)((m * n) & 127) / 128.f;
        dftRe[i] = (bf16)__cosf(th);
        dftIm[i] = (bf16)(-__sinf(th));
    }
    {   // inverse: [n][m], n<128, m<64 ; x_n = sum_m cM*(Re cos - Im sin)/128
        const int n = i >> 6, m = i & 63;
        const float c  = (m == 0 ? 1.f : 2.f) / 128.f;
        const float th = TWO_PI * (float)((m * n) & 127) / 128.f;
        idftC[i] = (bf16)(c * __cosf(th));
        idftS[i] = (bf16)(-c * __sinf(th));
    }
}

// ---------------------------------------------------------------------------
// Elementwise complex tanh: tanh(x+iy) = (sinh2x + i sin2y)/(cosh2x + cos2y)
// ---------------------------------------------------------------------------
__global__ void ctanh_kernel(const float* __restrict__ re, const float* __restrict__ im,
                             bf16* __restrict__ ore, bf16* __restrict__ oim, int n)
{
    const int i = blockIdx.x * blockDim.x + threadIdx.x;
    if (i >= n) return;
    float x = re[i], y = im[i];
    x = fminf(fmaxf(x, -15.f), 15.f);
    const float e2x = __expf(2.f * x), em2x = __expf(-2.f * x);
    const float sh = 0.5f * (e2x - em2x);
    const float ch = 0.5f * (e2x + em2x);
    float s, c;
    __sincosf(2.f * y, &s, &c);
    float d = ch + c;
    d = (fabsf(d) < 1e-30f) ? 1e-30f : d;
    ore[i] = (bf16)(sh / d);
    oim[i] = (bf16)(s / d);
}

// ---------------------------------------------------------------------------
// host-side launch helper
// ---------------------------------------------------------------------------
template<bool AF32, bool BF32, bool OUTBF16, bool OUTTRANS>
static void launch_gemm(dim3 grid,
                        const void* A, int lda, long sA,
                        const void* B, int ldb, long sB,
                        const void* A2, const void* B2,
                        const float* bias, const float* addin, long sAdd,
                        void* C, int ldc, long sC,
                        int K, int K2, float a1, float a2, hipStream_t s)
{
    gemm_nt<AF32, BF32, OUTBF16, OUTTRANS><<<grid, 256, 0, s>>>(
        A, B, A2, B2, bias, addin, C, K, K2, lda, ldb, ldc, sA, sB, sC, sAdd, a1, a2);
}

extern "C" void kernel_launch(void* const* d_in, const int* in_sizes, int n_in,
                              void* d_out, int out_size, void* d_ws, size_t ws_size,
                              hipStream_t stream)
{
    (void)in_sizes; (void)n_in; (void)out_size; (void)ws_size;

    const float* X      = (const float*)d_in[0]; // [64,128,1024]
    const float* W_lin  = (const float*)d_in[1]; // [512,1024]
    const float* b_lin  = (const float*)d_in[2]; // [512]
    const float* Wq     = (const float*)d_in[3]; // [512,512]
    const float* Wk     = (const float*)d_in[4]; // [512,512]
    const float* W_add  = (const float*)d_in[5]; // [512,512]
    const float* b_add  = (const float*)d_in[6]; // [512]
    const float* W_fin  = (const float*)d_in[7]; // [1024,65536]
    const float* b_fin  = (const float*)d_in[8]; // [1024]
    float* out = (float*)d_out;                  // [64,1024]

    char* ws = (char*)d_ws;
    // ---- workspace layout (bytes), all 256B aligned -----------------------
    const size_t O_DFTRE = 0;                    // [64][128] bf16
    const size_t O_DFTIM = O_DFTRE + 16384;
    const size_t O_IDFTC = O_DFTIM + 16384;      // [128][64] bf16
    const size_t O_IDFTS = O_IDFTC + 16384;
    const size_t O_EBF   = 65536;                // e  [8192][512] bf16 (-> reused for R)
    const size_t O_QT    = O_EBF + 8388608;      // qT [64][512][128] bf16 (-> Resid f32)
    const size_t O_KT    = O_QT  + 8388608;      // kT [64][512][128] bf16 (-> Resid f32)
    const size_t O_FQRE  = O_KT  + 8388608;      // [64][64][512] bf16 (-> FkT_re)
    const size_t O_FQIM  = O_FQRE + 4194304;     //                   (-> FkT_im)
    const size_t O_FKRE  = O_FQIM + 4194304;     //                   (-> XQT_re)
    const size_t O_FKIM  = O_FKRE + 4194304;     //                   (-> XQT_im)
    const size_t O_SRE   = O_FKIM + 4194304;     // [64][64][64] f32
    const size_t O_SIM   = O_SRE + 1048576;
    const size_t O_SPRE  = O_SIM + 1048576;      // [64][64][64] bf16
    const size_t O_SPIM  = O_SPRE + 524288;      // total ~43 MiB

    bf16* dftRe = (bf16*)(ws + O_DFTRE);
    bf16* dftIm = (bf16*)(ws + O_DFTIM);
    bf16* idftC = (bf16*)(ws + O_IDFTC);
    bf16* idftS = (bf16*)(ws + O_IDFTS);
    bf16* e_bf  = (bf16*)(ws + O_EBF);
    bf16* qT    = (bf16*)(ws + O_QT);
    bf16* kT    = (bf16*)(ws + O_KT);
    bf16* FqRe  = (bf16*)(ws + O_FQRE);
    bf16* FqIm  = (bf16*)(ws + O_FQIM);
    bf16* FkRe  = (bf16*)(ws + O_FKRE);
    bf16* FkIm  = (bf16*)(ws + O_FKIM);
    float* Sre  = (float*)(ws + O_SRE);
    float* Sim  = (float*)(ws + O_SIM);
    bf16* SpRe  = (bf16*)(ws + O_SPRE);
    bf16* SpIm  = (bf16*)(ws + O_SPIM);
    bf16* FkTRe = FqRe;                // reuse after scores
    bf16* FkTIm = FqIm;
    bf16* XQTre = FkRe;                // reuse after scores
    bf16* XQTim = FkIm;
    float* Resid = (float*)(ws + O_QT); // reuse qT+kT (16 MiB) after FkT
    bf16* R     = e_bf;                // reuse after Resid

    // 0) DFT tables
    init_tables<<<dim3(32), dim3(256), 0, stream>>>(dftRe, dftIm, idftC, idftS);

    // 1) e = X @ W_lin^T + b_lin  -> bf16 [8192,512]
    launch_gemm<true, true, true, false>(dim3(8, 128, 1),
        X, 1024, 0, W_lin, 1024, 0, nullptr, nullptr, b_lin, nullptr, 0,
        e_bf, 512, 0, 1024, 0, 1.f, 0.f, stream);

    // 2) qT[b,e,n] = (e_b @ Wq^T)^T ; kT likewise  (batched, transposed store)
    launch_gemm<false, true, true, true>(dim3(8, 2, 64),
        e_bf, 512, 65536, Wq, 512, 0, nullptr, nullptr, nullptr, nullptr, 0,
        qT, 128, 65536, 512, 0, 1.f, 0.f, stream);
    launch_gemm<false, true, true, true>(dim3(8, 2, 64),
        e_bf, 512, 65536, Wk, 512, 0, nullptr, nullptr, nullptr, nullptr, 0,
        kT, 128, 65536, 512, 0, 1.f, 0.f, stream);

    // 3) rfft via DFT matmul: F[m,e] = DFT[m,:] . qT[e,:]   (NT, K=128)
    launch_gemm<false, false, true, false>(dim3(8, 1, 64),
        dftRe, 128, 0, qT, 128, 65536, nullptr, nullptr, nullptr, nullptr, 0,
        FqRe, 512, 32768, 128, 0, 1.f, 0.f, stream);
    launch_gemm<false, false, true, false>(dim3(8, 1, 64),
        dftIm, 128, 0, qT, 128, 65536, nullptr, nullptr, nullptr, nullptr, 0,
        FqIm, 512, 32768, 128, 0, 1.f, 0.f, stream);
    launch_gemm<false, false, true, false>(dim3(8, 1, 64),
        dftRe, 128, 0, kT, 128, 65536, nullptr, nullptr, nullptr, nullptr, 0,
        FkRe, 512, 32768, 128, 0, 1.f, 0.f, stream);
    launch_gemm<false, false, true, false>(dim3(8, 1, 64),
        dftIm, 128, 0, kT, 128, 65536, nullptr, nullptr, nullptr, nullptr, 0,
        FkIm, 512, 32768, 128, 0, 1.f, 0.f, stream);

    // 4) scores (complex, no conjugate): S = Fq . Fk^T over e (K=512)
    launch_gemm<false, false, false, false>(dim3(1, 1, 64),
        FqRe, 512, 32768, FkRe, 512, 32768, FqIm, FkIm, nullptr, nullptr, 0,
        Sre, 64, 4096, 512, 512, 1.f, -1.f, stream);
    launch_gemm<false, false, false, false>(dim3(1, 1, 64),
        FqRe, 512, 32768, FkIm, 512, 32768, FqIm, FkRe, nullptr, nullptr, 0,
        Sim, 64, 4096, 512, 512, 1.f, 1.f, stream);

    // 5) complex tanh -> bf16
    ctanh_kernel<<<dim3((64*64*64 + 255)/256), dim3(256), 0, stream>>>(
        Sre, Sim, SpRe, SpIm, 64*64*64);

    // 6) FkT[b,e,m] (transposed DFT of k again; overwrites Fq slots)
    launch_gemm<false, false, true, true>(dim3(8, 1, 64),
        dftRe, 128, 0, kT, 128, 65536, nullptr, nullptr, nullptr, nullptr, 0,
        FkTRe, 64, 32768, 128, 0, 1.f, 0.f, stream);
    launch_gemm<false, false, true, true>(dim3(8, 1, 64),
        dftIm, 128, 0, kT, 128, 65536, nullptr, nullptr, nullptr, nullptr, 0,
        FkTIm, 64, 32768, 128, 0, 1.f, 0.f, stream);

    // 7) residual: Resid = e @ W_add^T + b_add  (f32, overwrites qT/kT)
    launch_gemm<false, true, false, false>(dim3(8, 2, 64),
        e_bf, 512, 65536, W_add, 512, 0, nullptr, nullptr, b_add, nullptr, 0,
        Resid, 512, 65536, 512, 0, 1.f, 0.f, stream);

    // 8) xqkv (complex): XQ[x,e] = sum_y S'[x,y] Fk[y,e], transposed bf16 out
    launch_gemm<false, false, true, true>(dim3(8, 1, 64),
        SpRe, 64, 4096, FkTRe, 64, 32768, SpIm, FkTIm, nullptr, nullptr, 0,
        XQTre, 64, 32768, 64, 64, 1.f, -1.f, stream);
    launch_gemm<false, false, true, true>(dim3(8, 1, 64),
        SpRe, 64, 4096, FkTIm, 64, 32768, SpIm, FkTRe, nullptr, nullptr, 0,
        XQTim, 64, 32768, 64, 64, 1.f, 1.f, stream);

    // 9) irfft via iDFT matmul + residual add, scaled by 1/(512*512):
    //    R[n,e] = a*(idftC . XQre^T + idftS . XQim^T) + Resid   -> bf16
    const float sc = 1.f / (512.f * 512.f);
    launch_gemm<false, false, true, false>(dim3(8, 2, 64),
        idftC, 64, 0, XQTre, 64, 32768, idftS, XQTim, nullptr, Resid, 65536,
        R, 512, 65536, 64, 64, sc, sc, stream);

    // 10) out = R_flat @ W_final^T + b_final  (M=64, N=1024, K=65536)
    launch_gemm<false, true, false, false>(dim3(16, 1, 1),
        R, 65536, 0, W_fin, 65536, 0, nullptr, nullptr, b_fin, nullptr, 0,
        out, 1024, 0, 65536, 0, 1.f, 0.f, stream);
}
